// RW_GNN_18202071400863
// MI455X (gfx1250) — matrix operand order repaired and checked by reading, not compiled, not run on hardware
//
#include <hip/hip_runtime.h>
#include <hip/hip_bf16.h>
#include <math.h>

#define N_NODES   200000
#define IN_DIM    128
#define HID       64
#define WALK_LEN  16
#define NUM_WALKS 16
#define N_GRAPHS  1024
#define PEN       32
#define N_CLASSES 10
#define BN_EPS    1e-5f

typedef __attribute__((ext_vector_type(16))) _Float16 v16h;
typedef __attribute__((ext_vector_type(8)))  float    v8f;

__device__ __forceinline__ v8f wmma_f16f32(v16h a, v16h b, v8f c) {
  // D = A(16x32 f16) x B(32x16 f16) + C(16x16 f32)
  return __builtin_amdgcn_wmma_f32_16x16x32_f16(false, a, false, b, (short)0, c, false, false);
}

// Load 16 f16 from two contiguous 8-element (16B) chunks -> A fragment halves.
__device__ __forceinline__ v16h ld16h_2x8(const _Float16* p0, const _Float16* p1) {
  union { uint4 q[2]; v16h v; } u;
  u.q[0] = *(const uint4*)p0;
  u.q[1] = *(const uint4*)p1;
  return u.v;
}

// Load 16 contiguous f16 (32B) -> B fragment.
__device__ __forceinline__ v16h ld16h(const _Float16* p) {
  union { uint4 q[2]; v16h v; } u;
  u.q[0] = ((const uint4*)p)[0];
  u.q[1] = ((const uint4*)p)[1];
  return u.v;
}

__device__ __forceinline__ float sigmoidf_(float x) { return 1.0f / (1.0f + expf(-x)); }

// A-fragment gather for one walk step: two 16B chunks per k-tile from the
// (L2-resident) f16 encoded-feature table.
__device__ __forceinline__ void load_ax(const _Float16* __restrict__ xench, int node,
                                        int khalf, v16h ax[2]) {
  const _Float16* xr = xench + (size_t)node * HID;
  #pragma unroll
  for (int kk = 0; kk < 2; kk++) {
    const int kb = kk * 32 + khalf * 16;
    ax[kk] = ld16h_2x8(xr + kb, xr + kb + 8);
  }
}

// ---------------------------------------------------------------------------
// Kernel 1: x_enc = f16( x @ W_enc^T + b_enc ), stored row-major [N_NODES][64].
// One wave per 16-node tile; K=128 as 4 WMMA k-steps; N=64 as 4 tiles.
// ---------------------------------------------------------------------------
__global__ __launch_bounds__(256)
void encode_kernel(const float* __restrict__ x, const float* __restrict__ Wenc,
                   const float* __restrict__ benc, _Float16* __restrict__ xench) {
  __shared__ __align__(16) _Float16 sW[HID * IN_DIM];  // W_enc as f16, row-major [h][k]
  __shared__ float sB[HID];
  const int tid = threadIdx.x;
  for (int i = tid; i < HID * IN_DIM; i += 256) sW[i] = (_Float16)Wenc[i];
  if (tid < HID) sB[tid] = benc[tid];
  __syncthreads();

  const int wave = tid >> 5, lane = tid & 31;
  const int row = lane & 15, khalf = lane >> 4, col = lane & 15;
  const int m0 = (blockIdx.x * 8 + wave) * 16;
  int node = m0 + row;
  if (node >= N_NODES) node = N_NODES - 1;   // clamp loads; stores guarded below
  const float* xr = x + (size_t)node * IN_DIM;

  v8f acc[4] = {};
  #pragma unroll
  for (int kk = 0; kk < 4; kk++) {
    const int kb = kk * 32 + khalf * 16;
    float t[16];
    *(float4*)(t + 0)  = *(const float4*)(xr + kb);
    *(float4*)(t + 4)  = *(const float4*)(xr + kb + 4);
    *(float4*)(t + 8)  = *(const float4*)(xr + kb + 8);
    *(float4*)(t + 12) = *(const float4*)(xr + kb + 12);
    v16h a;
    #pragma unroll
    for (int e = 0; e < 16; e++) a[e] = (_Float16)t[e];
    #pragma unroll
    for (int nt = 0; nt < 4; nt++) {
      const _Float16* bp = sW + (nt * 16 + col) * IN_DIM + kk * 32 + khalf * 16;
      acc[nt] = wmma_f16f32(a, ld16h(bp), acc[nt]);
    }
  }
  #pragma unroll
  for (int nt = 0; nt < 4; nt++) {
    #pragma unroll
    for (int v = 0; v < 8; v++) {
      const int m = m0 + v + khalf * 8;     // C layout: VGPR v holds rows v / v+8
      if (m < N_NODES)
        xench[(size_t)m * HID + nt * 16 + col] = (_Float16)(acc[nt][v] + sB[nt * 16 + col]);
    }
  }
}

// ---------------------------------------------------------------------------
// Kernel 2: fused gather + 16-step GRU + per-graph mean. One wave == one graph
// (16 walks == WMMA M). r/z gates fuse x- and h-GEMMs into one accumulator.
// The walk-feature gather for step t+1 is software-pipelined past step t's
// WMMA + gate-math body to hide L2 latency on the recurrent chain.
// ---------------------------------------------------------------------------
__global__ __launch_bounds__(128)
void gru_pool_kernel(const _Float16* __restrict__ xench, const int* __restrict__ walks,
                     const float* __restrict__ Wih, const float* __restrict__ Whh,
                     const float* __restrict__ bih, const float* __restrict__ bhh,
                     float* __restrict__ graph_enc) {
  __shared__ __align__(16) _Float16 sWih[3 * HID * HID];   // f16 row-major [192][64]
  __shared__ __align__(16) _Float16 sWhh[3 * HID * HID];
  __shared__ __align__(16) _Float16 sH[4][16 * HID];       // per-wave h staging (f16)
  __shared__ __align__(16) int      sIdx[4][NUM_WALKS * WALK_LEN];
  __shared__ float sBih[3 * HID], sBhh[3 * HID];

  const int tid = threadIdx.x;
  for (int i = tid; i < 3 * HID * HID; i += 128) {
    sWih[i] = (_Float16)Wih[i];
    sWhh[i] = (_Float16)Whh[i];
  }
  for (int i = tid; i < 3 * HID; i += 128) { sBih[i] = bih[i]; sBhh[i] = bhh[i]; }

  const int wave = tid >> 5, lane = tid & 31;
  const int g = blockIdx.x * 4 + wave;
  {  // stage this graph's 256 walk indices (coalesced b128)
    const int4* src = (const int4*)(walks + (size_t)g * NUM_WALKS * WALK_LEN);
    int4* dst = (int4*)sIdx[wave];
    dst[lane * 2 + 0] = src[lane * 2 + 0];
    dst[lane * 2 + 1] = src[lane * 2 + 1];
  }
  {  // zero h staging tile (h0 = 0)
    uint4 z = {0u, 0u, 0u, 0u};
    uint4* hz = (uint4*)sH[wave];
    hz[lane * 4 + 0] = z; hz[lane * 4 + 1] = z;
    hz[lane * 4 + 2] = z; hz[lane * 4 + 3] = z;
  }
  __syncthreads();

  const int row = lane & 15, khalf = lane >> 4, col = lane & 15;
  float hcur[4][8];
  #pragma unroll
  for (int j = 0; j < 4; j++)
    #pragma unroll
    for (int v = 0; v < 8; v++) hcur[j][v] = 0.f;

  // prologue: gather step 0's features
  v16h ax[2], axn[2];
  load_ax(xench, sIdx[wave][row * WALK_LEN + 0], khalf, ax);

  for (int t = 0; t < WALK_LEN; t++) {
    // issue next step's gather early; consumed only at the iteration boundary
    if (t + 1 < WALK_LEN)
      load_ax(xench, sIdx[wave][row * WALK_LEN + t + 1], khalf, axn);

    v16h ah[2];
    #pragma unroll
    for (int kk = 0; kk < 2; kk++) {
      const int kb = kk * 32 + khalf * 16;
      ah[kk] = ld16h_2x8(&sH[wave][row * HID + kb], &sH[wave][row * HID + kb + 8]);
    }
    v8f ar[4] = {}, az[4] = {}, an_i[4] = {}, an_h[4] = {};
    #pragma unroll
    for (int j = 0; j < 4; j++) {
      #pragma unroll
      for (int kk = 0; kk < 2; kk++) {
        const int off = kk * 32 + khalf * 16;
        ar[j]   = wmma_f16f32(ax[kk], ld16h(sWih + (       j * 16 + col) * HID + off), ar[j]);
        ar[j]   = wmma_f16f32(ah[kk], ld16h(sWhh + (       j * 16 + col) * HID + off), ar[j]);
        az[j]   = wmma_f16f32(ax[kk], ld16h(sWih + ( 64 +  j * 16 + col) * HID + off), az[j]);
        az[j]   = wmma_f16f32(ah[kk], ld16h(sWhh + ( 64 +  j * 16 + col) * HID + off), az[j]);
        an_i[j] = wmma_f16f32(ax[kk], ld16h(sWih + (128 +  j * 16 + col) * HID + off), an_i[j]);
        an_h[j] = wmma_f16f32(ah[kk], ld16h(sWhh + (128 +  j * 16 + col) * HID + off), an_h[j]);
      }
    }
    #pragma unroll
    for (int j = 0; j < 4; j++) {
      const int ic = j * 16 + col;
      const float br  = sBih[ic] + sBhh[ic];
      const float bz  = sBih[64 + ic] + sBhh[64 + ic];
      const float bni = sBih[128 + ic], bnh = sBhh[128 + ic];
      #pragma unroll
      for (int v = 0; v < 8; v++) {
        const float r = sigmoidf_(ar[j][v] + br);
        const float z = sigmoidf_(az[j][v] + bz);
        const float n = tanhf(an_i[j][v] + bni + r * (an_h[j][v] + bnh));
        hcur[j][v] = (1.0f - z) * n + z * hcur[j][v];
      }
    }
    // write h back as f16 A-source for next step (wave-private region)
    #pragma unroll
    for (int j = 0; j < 4; j++)
      #pragma unroll
      for (int v = 0; v < 8; v++)
        sH[wave][(v + khalf * 8) * HID + j * 16 + col] = (_Float16)hcur[j][v];

    // rotate pipelined gather
    ax[0] = axn[0];
    ax[1] = axn[1];
  }

  // mean over the graph's 16 walks: sum 8 rows in-register + cross-half shuffle
  #pragma unroll
  for (int j = 0; j < 4; j++) {
    float s = 0.f;
    #pragma unroll
    for (int v = 0; v < 8; v++) s += hcur[j][v];
    s += __shfl_xor(s, 16, 32);
    if (lane < 16)
      graph_enc[(size_t)g * HID + j * 16 + lane] = s * (1.0f / NUM_WALKS);
  }
}

// ---------------------------------------------------------------------------
// Kernel 3: BN batch stats (mean, biased var) per feature.
// ---------------------------------------------------------------------------
__global__ __launch_bounds__(64)
void bn_stats_kernel(const float* __restrict__ ge, float* __restrict__ stats) {
  const int c = threadIdx.x;
  float s = 0.f, s2 = 0.f;
  for (int gi = 0; gi < N_GRAPHS; gi++) {
    const float v = ge[(size_t)gi * HID + c];
    s += v; s2 += v * v;
  }
  const float m = s * (1.0f / N_GRAPHS);
  stats[c] = m;
  stats[HID + c] = s2 * (1.0f / N_GRAPHS) - m * m;
}

// ---------------------------------------------------------------------------
// Kernel 4: BN apply + MLP head + log_softmax. One thread per graph.
// ---------------------------------------------------------------------------
__global__ __launch_bounds__(256)
void head_kernel(const float* __restrict__ ge, const float* __restrict__ stats,
                 const float* __restrict__ gamma, const float* __restrict__ beta,
                 const float* __restrict__ W1, const float* __restrict__ b1,
                 const float* __restrict__ W2, const float* __restrict__ b2,
                 float* __restrict__ out) {
  __shared__ float sW1[PEN * HID], sW2[N_CLASSES * PEN], sb1[PEN], sb2[N_CLASSES];
  __shared__ float sScale[HID], sShift[HID];
  const int tid = threadIdx.x;
  for (int i = tid; i < PEN * HID; i += 256) sW1[i] = W1[i];
  if (tid < N_CLASSES * PEN) sW2[tid] = W2[tid];
  if (tid < PEN) sb1[tid] = b1[tid];
  if (tid < N_CLASSES) sb2[tid] = b2[tid];
  if (tid < HID) {
    const float inv = rsqrtf(stats[HID + tid] + BN_EPS);
    const float sc = gamma[tid] * inv;
    sScale[tid] = sc;
    sShift[tid] = beta[tid] - stats[tid] * sc;
  }
  __syncthreads();
  const int g = blockIdx.x * 256 + tid;
  if (g >= N_GRAPHS) return;

  float gn[HID];
  for (int k = 0; k < HID; k++) gn[k] = ge[(size_t)g * HID + k] * sScale[k] + sShift[k];
  float h1[PEN];
  for (int p = 0; p < PEN; p++) {
    float a = sb1[p];
    for (int k = 0; k < HID; k++) a += sW1[p * HID + k] * gn[k];
    h1[p] = a > 0.f ? a : 0.f;
  }
  float lg[N_CLASSES];
  float mx = -1e30f;
  for (int c = 0; c < N_CLASSES; c++) {
    float a = sb2[c];
    for (int p = 0; p < PEN; p++) a += sW2[c * PEN + p] * h1[p];
    lg[c] = a;
    mx = fmaxf(mx, a);
  }
  float se = 0.f;
  for (int c = 0; c < N_CLASSES; c++) se += expf(lg[c] - mx);
  const float lse = mx + logf(se);
  for (int c = 0; c < N_CLASSES; c++) out[(size_t)g * N_CLASSES + c] = lg[c] - lse;
}

// ---------------------------------------------------------------------------
extern "C" void kernel_launch(void* const* d_in, const int* in_sizes, int n_in,
                              void* d_out, int out_size, void* d_ws, size_t ws_size,
                              hipStream_t stream) {
  (void)in_sizes; (void)n_in; (void)out_size; (void)ws_size;
  const float* x     = (const float*)d_in[0];
  const int*   walks = (const int*)d_in[1];
  // d_in[2] walk_batch == repeat(arange(N_GRAPHS), 16): consecutive-16 grouping
  // is baked into the GRU kernel's wave-per-graph mapping.
  const float* Wenc  = (const float*)d_in[3];
  const float* benc  = (const float*)d_in[4];
  const float* Wih   = (const float*)d_in[5];
  const float* Whh   = (const float*)d_in[6];
  const float* bih   = (const float*)d_in[7];
  const float* bhh   = (const float*)d_in[8];
  const float* gamma = (const float*)d_in[9];
  const float* beta  = (const float*)d_in[10];
  const float* W1    = (const float*)d_in[11];
  const float* b1    = (const float*)d_in[12];
  const float* W2    = (const float*)d_in[13];
  const float* b2    = (const float*)d_in[14];
  float* out = (float*)d_out;

  char* ws = (char*)d_ws;
  _Float16* xench     = (_Float16*)ws;                                  // 25.6 MB (L2-resident)
  float*    graph_enc = (float*)(ws + (size_t)N_NODES * HID * sizeof(_Float16));
  float*    stats     = graph_enc + (size_t)N_GRAPHS * HID;

  encode_kernel<<<(N_NODES + 127) / 128, 256, 0, stream>>>(x, Wenc, benc, xench);
  gru_pool_kernel<<<N_GRAPHS / 4, 128, 0, stream>>>(xench, walks, Wih, Whh, bih, bhh, graph_enc);
  bn_stats_kernel<<<1, 64, 0, stream>>>(graph_enc, stats);
  head_kernel<<<(N_GRAPHS + 255) / 256, 256, 0, stream>>>(graph_enc, stats, gamma, beta,
                                                          W1, b1, W2, b2, out);
}
